// SiameseNetwork_58308476011182
// MI455X (gfx1250) — matrix operand order repaired and checked by reading, not compiled, not run on hardware
//
#include <hip/hip_runtime.h>
#include <hip/hip_bf16.h>
#include <cstdint>

#define VOCAB  50000
#define EMBED  300
#define KPAD   320            // EMBED zero-padded to a multiple of 32
#define HIDDEN 512
#define GATES  (4 * HIDDEN)   // 2048
#define BATCH  128
#define SEQ    256
#define HPAD   520            // padded LDS row stride (bf16 elems), 16B-multiple, bank-friendly

typedef __attribute__((ext_vector_type(16))) __bf16   v16bf;
typedef __attribute__((ext_vector_type(8)))  float    v8f;
typedef __attribute__((ext_vector_type(4)))  uint32_t u32x4;
typedef __attribute__((ext_vector_type(4)))  int      v4i;

union FragBF {           // one WMMA bf16 operand fragment (32 bytes / lane)
    u32x4 u[2];
    v16bf v;
};

__device__ __forceinline__ unsigned short f32_to_bf16(float f) {
    uint32_t u = __float_as_uint(f);
    uint32_t r = u + 0x7FFFu + ((u >> 16) & 1u);   // round-to-nearest-even
    return (unsigned short)(r >> 16);
}
__device__ __forceinline__ float bf16_to_f32(uint32_t h) {
    return __uint_as_float(h << 16);
}
__device__ __forceinline__ float sigf(float x)      { return 1.0f / (1.0f + __expf(-x)); }
__device__ __forceinline__ float tanhfast(float x)  { return 2.0f * sigf(2.0f * x) - 1.0f; }

// ---- async global->LDS (CDNA5 ASYNCcnt path), guarded for toolchain drift ----
#if __has_builtin(__builtin_amdgcn_global_load_async_to_lds_b128)
#define HAVE_ASYNC_LDS 1
#define GLOBAL_AS __attribute__((address_space(1)))
#define LDS_AS    __attribute__((address_space(3)))
__device__ __forceinline__ void async_copy16(const unsigned short* g, unsigned short* l) {
    __builtin_amdgcn_global_load_async_to_lds_b128(
        (GLOBAL_AS v4i*)g, (LDS_AS v4i*)l, 0, 0);
}
__device__ __forceinline__ void wait_async0() {
#if __has_builtin(__builtin_amdgcn_s_wait_asynccnt)
    __builtin_amdgcn_s_wait_asynccnt(0);
#else
    asm volatile("s_wait_asynccnt 0x0" ::: "memory");
#endif
}
#else
#define HAVE_ASYNC_LDS 0
#endif

// ---------------------------------------------------------------------------
// Kernel 0: fp32 -> bf16 table conversion with K padding.
//   out[row*KP + col] = col < K ? bf16(in[row*K + col]) : 0
// Used for emb (300->320), W_ih (300->320), W_hh (512->512).
// ---------------------------------------------------------------------------
__global__ __launch_bounds__(256) void k_cvt_pad(const float* __restrict__ in,
                                                 unsigned short* __restrict__ out,
                                                 int rows, int K, int KP) {
    long i = (long)blockIdx.x * 256 + threadIdx.x;
    long total = (long)rows * KP;
    if (i >= total) return;
    int row = (int)(i / KP), col = (int)(i % KP);
    float v = (col < K) ? in[(size_t)row * K + col] : 0.0f;
    out[i] = f32_to_bf16(v);
}

// ---------------------------------------------------------------------------
// Kernel 1: embedding lookup + input projection (bf16 tables, no conversion).
//   xw[q][t][n][b] = bf16( emb[tok[b,t]] . W_ih[n] + b_ih[n] + b_hh[n] )
// Grid: (SEQ, GATES/128, 2). One block = one t, one 128-wide N tile, one query.
// Tiles staged into LDS with per-lane async global->LDS b128 copies
// (the A tile is a token-indexed gather: each lane supplies its own address).
// ---------------------------------------------------------------------------
__global__ __launch_bounds__(256) void k_embed_proj(
    const int* __restrict__ q1, const int* __restrict__ q2,
    const unsigned short* __restrict__ emb_bf,
    const unsigned short* __restrict__ wih_bf,
    const float* __restrict__ b_ih, const float* __restrict__ b_hh,
    unsigned short* __restrict__ xw)
{
    const int t  = blockIdx.x;
    const int nb = blockIdx.y;
    const int qi = blockIdx.z;
    const int* q = qi ? q2 : q1;
    unsigned short* xwq = xw + (size_t)qi * SEQ * GATES * BATCH;

    const int tid    = threadIdx.x;
    const int wave   = tid >> 5;
    const int lane   = tid & 31;
    const int nlo    = lane & 15;
    const int hiHalf = (lane >= 16);

    __shared__ int tok[BATCH];
    __shared__ unsigned short As[BATCH * 40];  // [b][40] bf16, 32 cols used
    __shared__ unsigned short Bs[128 * 40];    // [n][40] bf16

    if (tid < BATCH) tok[tid] = q[tid * SEQ + t];

    v8f acc[8];
    #pragma unroll
    for (int i = 0; i < 8; ++i)
        #pragma unroll
        for (int j = 0; j < 8; ++j) acc[i][j] = 0.0f;

    __syncthreads();

    for (int k0 = 0; k0 < KPAD; k0 += 32) {
        // ---- stage A (gathered emb rows) and B (W_ih) tiles: 16B chunks ----
        #pragma unroll
        for (int cc = 0; cc < 2; ++cc) {
            int c   = tid + cc * 256;      // 512 chunks per tile
            int row = c >> 2;
            int sub = (c & 3) * 8;         // bf16 element offset within 32-col slab
            const unsigned short* ga = emb_bf + (size_t)tok[row] * KPAD + k0 + sub;
            const unsigned short* gb = wih_bf + (size_t)(nb * 128 + row) * KPAD + k0 + sub;
            unsigned short* la = &As[row * 40 + sub];
            unsigned short* lb = &Bs[row * 40 + sub];
#if HAVE_ASYNC_LDS
            async_copy16(ga, la);
            async_copy16(gb, lb);
#else
            *(u32x4*)la = *(const u32x4*)ga;
            *(u32x4*)lb = *(const u32x4*)gb;
#endif
        }
#if HAVE_ASYNC_LDS
        wait_async0();
#endif
        __syncthreads();

        // ---- A fragment (16x32, per ISA layout) ----
        const int mrow = wave * 16 + nlo;
        const int ka   = hiHalf ? 8 : 0;
        FragBF a;
        a.u[0] = *(const u32x4*)&As[mrow * 40 + ka];
        a.u[1] = *(const u32x4*)&As[mrow * 40 + ka + 16];

        #pragma unroll
        for (int nt = 0; nt < 8; ++nt) {
            const int ncol = nt * 16 + nlo;
            const int kb   = hiHalf ? 8 : 0;
            FragBF b;
            b.u[0] = *(const u32x4*)&Bs[ncol * 40 + kb];
            b.u[1] = *(const u32x4*)&Bs[ncol * 40 + kb + 16];
            acc[nt] = __builtin_amdgcn_wmma_f32_16x16x32_bf16(
                false, a.v, false, b.v, (short)0, acc[nt], false, false);
        }
        __syncthreads();
    }

    // ---- epilogue: add biases, pack bf16x8, store one b128 per tile ----
    const int b0 = wave * 16 + (hiHalf ? 8 : 0);
    #pragma unroll
    for (int nt = 0; nt < 8; ++nt) {
        int n = nb * 128 + nt * 16 + nlo;
        float bias = b_ih[n] + b_hh[n];
        uint32_t p[4];
        #pragma unroll
        for (int k = 0; k < 4; ++k) {
            uint32_t lo = f32_to_bf16(acc[nt][2 * k]     + bias);
            uint32_t hi = f32_to_bf16(acc[nt][2 * k + 1] + bias);
            p[k] = lo | (hi << 16);
        }
        u32x4 st = { p[0], p[1], p[2], p[3] };
        size_t off = ((size_t)t * GATES + n) * BATCH + b0;
        *(u32x4*)&xwq[off] = st;
    }
}

// ---------------------------------------------------------------------------
// Kernel 2: LSTM recurrence. Grid: 16 blocks (2 queries x 8 batch tiles of 16).
// 256 threads = 8 waves; wave w owns hidden slice [w*64, w*64+64) -> for each
// gate block g in {i,f,g,o} the 4 N-subtiles g*512 + w*64 + s*16. All four
// gates of a hidden unit land in the same lane slot -> in-register gate math.
// h lives in LDS (bf16, padded rows); c stays in VGPRs.
// W_hh (bf16, 2MB) streamed from L2 every step.
// ---------------------------------------------------------------------------
__global__ __launch_bounds__(256) void k_lstm(
    const unsigned short* __restrict__ xw,
    const unsigned short* __restrict__ whh_bf,
    const int* __restrict__ len1, const int* __restrict__ len2,
    float* __restrict__ hfinal)
{
    const int qi = blockIdx.x >> 3;
    const int bt = blockIdx.x & 7;
    const int b0 = bt * 16;
    const int* lens = qi ? len2 : len1;
    const unsigned short* xwq = xw + (size_t)qi * SEQ * GATES * BATCH;
    float* hfq = hfinal + (size_t)qi * BATCH * HIDDEN;

    const int tid    = threadIdx.x;
    const int wave   = tid >> 5;
    const int lane   = tid & 31;
    const int nlo    = lane & 15;
    const int hiHalf = (lane >= 16);

    __shared__ unsigned short hbf[16 * HPAD];
    for (int i = tid; i < 16 * HPAD; i += 256) hbf[i] = 0;

    int len_m[8];
    #pragma unroll
    for (int r = 0; r < 8; ++r)
        len_m[r] = lens[b0 + (hiHalf ? 8 : 0) + r];

    v8f c[4];
    #pragma unroll
    for (int s = 0; s < 4; ++s)
        #pragma unroll
        for (int r = 0; r < 8; ++r) c[s][r] = 0.0f;

    __syncthreads();

    for (int t = 0; t < SEQ; ++t) {
        // ---- init accumulators from precomputed xw (C-layout-matched b128) ----
        v8f acc[16];
        #pragma unroll
        for (int g = 0; g < 4; ++g)
            #pragma unroll
            for (int s = 0; s < 4; ++s) {
                int n = g * HIDDEN + wave * 64 + s * 16 + nlo;
                size_t off = ((size_t)t * GATES + n) * BATCH + b0 + (hiHalf ? 8 : 0);
                u32x4 u = *(const u32x4*)&xwq[off];
                v8f a;
                a[0] = bf16_to_f32(u.x & 0xFFFFu); a[1] = bf16_to_f32(u.x >> 16);
                a[2] = bf16_to_f32(u.y & 0xFFFFu); a[3] = bf16_to_f32(u.y >> 16);
                a[4] = bf16_to_f32(u.z & 0xFFFFu); a[5] = bf16_to_f32(u.z >> 16);
                a[6] = bf16_to_f32(u.w & 0xFFFFu); a[7] = bf16_to_f32(u.w >> 16);
                acc[g * 4 + s] = a;
            }

        // prefetch next step's xw slab while we grind the GEMM
        if (t + 1 < SEQ) {
            int n = wave * 64 + nlo;
            size_t off = ((size_t)(t + 1) * GATES + n) * BATCH + b0;
            __builtin_prefetch(&xwq[off], 0, 1);
        }

        // ---- gates += h_{t-1} @ W_hh^T  (K = 512, 16 k-steps of 32) ----
        for (int k0 = 0; k0 < HIDDEN; k0 += 32) {
            const int ka = hiHalf ? 8 : 0;
            FragBF a;                                  // A = h tile 16x32
            a.u[0] = *(const u32x4*)&hbf[nlo * HPAD + k0 + ka];
            a.u[1] = *(const u32x4*)&hbf[nlo * HPAD + k0 + ka + 16];
            const int kb = k0 + (hiHalf ? 8 : 0);
            #pragma unroll
            for (int g = 0; g < 4; ++g)
                #pragma unroll
                for (int s = 0; s < 4; ++s) {
                    int ncol = g * HIDDEN + wave * 64 + s * 16 + nlo;
                    const unsigned short* wp = whh_bf + (size_t)ncol * HIDDEN + kb;
                    FragBF b;
                    b.u[0] = *(const u32x4*)wp;
                    b.u[1] = *(const u32x4*)(wp + 16);
                    acc[g * 4 + s] = __builtin_amdgcn_wmma_f32_16x16x32_bf16(
                        false, a.v, false, b.v, (short)0, acc[g * 4 + s], false, false);
                }
        }

        __syncthreads();   // all waves done reading h_{t-1} from LDS

        // ---- gate nonlinearity, c/h update, export h_t to LDS ----
        #pragma unroll
        for (int s = 0; s < 4; ++s) {
            v8f vi = acc[0 * 4 + s], vf = acc[1 * 4 + s];
            v8f vg = acc[2 * 4 + s], vo = acc[3 * 4 + s];
            int j = wave * 64 + s * 16 + nlo;
            #pragma unroll
            for (int r = 0; r < 8; ++r) {
                float cn = sigf(vf[r]) * c[s][r] + sigf(vi[r]) * tanhfast(vg[r]);
                c[s][r] = cn;
                float h = sigf(vo[r]) * tanhfast(cn);
                int m = (hiHalf ? 8 : 0) + r;
                hbf[m * HPAD + j] = f32_to_bf16(h);
                if (t + 1 == len_m[r])
                    hfq[(size_t)(b0 + m) * HIDDEN + j] = h;   // last valid state
            }
        }
        __syncthreads();   // h_t visible before next step's A-fragment reads
    }
}

// ---------------------------------------------------------------------------
// Kernel 3: out[b] = exp(-sum_j |h1[b,j] - h2[b,j]|). 128 blocks x 64 threads.
// ---------------------------------------------------------------------------
__global__ __launch_bounds__(64) void k_dist(const float* __restrict__ hfinal,
                                             float* __restrict__ out) {
    int b   = blockIdx.x;
    int tid = threadIdx.x;
    const float* h1 = hfinal + (size_t)b * HIDDEN;
    const float* h2 = hfinal + (size_t)(BATCH + b) * HIDDEN;
    float s = 0.0f;
    for (int j = tid; j < HIDDEN; j += 64) s += fabsf(h1[j] - h2[j]);
    #pragma unroll
    for (int off = 16; off > 0; off >>= 1) s += __shfl_xor(s, off, 32);
    __shared__ float red[2];
    if ((tid & 31) == 0) red[tid >> 5] = s;
    __syncthreads();
    if (tid == 0) out[b] = __expf(-(red[0] + red[1]));
}

// ---------------------------------------------------------------------------
extern "C" void kernel_launch(void* const* d_in, const int* in_sizes, int n_in,
                              void* d_out, int out_size, void* d_ws, size_t ws_size,
                              hipStream_t stream) {
    (void)in_sizes; (void)n_in; (void)out_size; (void)ws_size;
    const int*   q1   = (const int*)  d_in[0];
    const int*   q2   = (const int*)  d_in[1];
    const int*   l1   = (const int*)  d_in[2];
    const int*   l2   = (const int*)  d_in[3];
    const float* emb  = (const float*)d_in[4];
    const float* W_ih = (const float*)d_in[5];
    const float* W_hh = (const float*)d_in[6];
    const float* b_ih = (const float*)d_in[7];
    const float* b_hh = (const float*)d_in[8];
    float* out = (float*)d_out;

    // workspace layout:
    //  [whh_bf 2MB][wih_bf 1.25MB][emb_bf 30.5MB][xw bf16 256MB][hfinal f32 512KB]
    char* ws = (char*)d_ws;
    unsigned short* whh_bf = (unsigned short*)ws;
    size_t off = (size_t)GATES * HIDDEN * 2;
    unsigned short* wih_bf = (unsigned short*)(ws + off);
    off += (size_t)GATES * KPAD * 2;
    unsigned short* emb_bf = (unsigned short*)(ws + off);
    off += (size_t)VOCAB * KPAD * 2;
    unsigned short* xw = (unsigned short*)(ws + off);
    off += (size_t)2 * SEQ * GATES * BATCH * 2;
    float* hfinal = (float*)(ws + off);

    long n_whh = (long)GATES * HIDDEN;
    long n_wih = (long)GATES * KPAD;
    long n_emb = (long)VOCAB * KPAD;
    k_cvt_pad<<<dim3((unsigned)((n_whh + 255) / 256)), dim3(256), 0, stream>>>(
        W_hh, whh_bf, GATES, HIDDEN, HIDDEN);
    k_cvt_pad<<<dim3((unsigned)((n_wih + 255) / 256)), dim3(256), 0, stream>>>(
        W_ih, wih_bf, GATES, EMBED, KPAD);
    k_cvt_pad<<<dim3((unsigned)((n_emb + 255) / 256)), dim3(256), 0, stream>>>(
        emb, emb_bf, VOCAB, EMBED, KPAD);

    k_embed_proj<<<dim3(SEQ, GATES / 128, 2), dim3(256), 0, stream>>>(
        q1, q2, emb_bf, wih_bf, b_ih, b_hh, xw);
    k_lstm<<<dim3(16), dim3(256), 0, stream>>>(xw, whh_bf, l1, l2, hfinal);
    k_dist<<<dim3(BATCH), dim3(64), 0, stream>>>(hfinal, out);
}
